// FEDFormer_11811160064536
// MI455X (gfx1250) — compile-verified, matrix operand order checked by reading, not run on hardware
//
#include <hip/hip_runtime.h>
#include <hip/hip_bf16.h>
#include <math.h>

typedef __attribute__((ext_vector_type(16))) _Float16 v16h;
typedef __attribute__((ext_vector_type(8)))  _Float16 v8h;
typedef __attribute__((ext_vector_type(8)))  float    v8f;
typedef _Float16 h16;

#define NB    32
#define LSEQ  512
#define DM    512
#define NH    8
#define DKH   64
#define DFF   2048
#define NMOD  64
#define CINC  7
#define NTIME 4
#define PRED  256
#define ROWS  (NB*LSEQ)           // 16384
#define LN_EPS 1e-5f

#define GF_BIAS 1
#define GF_GELU 2
#define GF_RES  4

#define BMT 256
#define BNT 64
#define BKT 32

// ---------------------------------------------------------------------------
// WMMA GEMM (all-f16, exact tiling: M%256==0, N%64==0, K%32==0)
//   C[M,N](f16) = A[M,K](f16) * Bt[N,K](f16, pre-transposed)  (+bias/gelu/res)
// Double-buffered LDS; tiles staged with async global->LDS B128 copies that
// overlap the WMMA work of the previous tile (tracked by ASYNCcnt).
// Wave tile 64x32 = 8 x v_wmma_f32_16x16x32_f16 per K-step.
// ---------------------------------------------------------------------------
template<int FLAGS>
__global__ __launch_bounds__(256)
void k_wmma_gemm(const h16* __restrict__ A, const h16* __restrict__ Bt,
                 h16* __restrict__ C, const float* __restrict__ bias,
                 const h16* __restrict__ resid, int M, int N, int K)
{
    __shared__ __align__(16) h16 As[2][BMT * BKT];   // 2 x 16 KB
    __shared__ __align__(16) h16 Bs[2][BNT * BKT];   // 2 x 4 KB

    const int tid  = threadIdx.x;
    const int wave = tid >> 5;
    const int lane = tid & 31;
    const int wm   = wave & 3;     // 4 waves along M (64 rows each)
    const int wn   = wave >> 2;    // 2 waves along N (32 cols each)
    const int lm   = lane & 15;
    const int lh   = lane >> 4;

    const int m0 = blockIdx.y * BMT;
    const int n0 = blockIdx.x * BNT;

    // staging addresses: thread stages one full 64B A-row and one 16B B-chunk
    const unsigned lA[2] = {
        (unsigned)(uintptr_t)((char*)As[0] + tid * 64),
        (unsigned)(uintptr_t)((char*)As[1] + tid * 64) };
    const unsigned lB[2] = {
        (unsigned)(uintptr_t)((char*)Bs[0] + (tid >> 2) * 64 + ((tid & 3) << 4)),
        (unsigned)(uintptr_t)((char*)Bs[1] + (tid >> 2) * 64 + ((tid & 3) << 4)) };
    const size_t aRow = (size_t)(m0 + tid) * K;
    const size_t bRow = (size_t)(n0 + (tid >> 2)) * K;

#define STAGE(buf, k0)                                                                   \
    {                                                                                    \
        const char* ga = (const char*)A + ((aRow + (k0)) << 1);                          \
        const char* gb = (const char*)Bt + ((bRow + (k0)) << 1) + ((tid & 3) << 4);      \
        asm volatile("global_load_async_to_lds_b128 %0, %1, off"           :: "v"(lA[buf]), "v"(ga) : "memory"); \
        asm volatile("global_load_async_to_lds_b128 %0, %1, off offset:16" :: "v"(lA[buf]), "v"(ga) : "memory"); \
        asm volatile("global_load_async_to_lds_b128 %0, %1, off offset:32" :: "v"(lA[buf]), "v"(ga) : "memory"); \
        asm volatile("global_load_async_to_lds_b128 %0, %1, off offset:48" :: "v"(lA[buf]), "v"(ga) : "memory"); \
        asm volatile("global_load_async_to_lds_b128 %0, %1, off"           :: "v"(lB[buf]), "v"(gb) : "memory"); \
    }

    v8f acc[4][2] = {};

    STAGE(0, 0)
    asm volatile("s_wait_asynccnt 0x0" ::: "memory");
    __syncthreads();

    const int nk = K >> 5;
    for (int ki = 0; ki < nk; ++ki) {
        const int buf = ki & 1;
        if (ki + 1 < nk) STAGE(buf ^ 1, (ki + 1) << 5)

        v16h bf[2];
#pragma unroll
        for (int nt = 0; nt < 2; ++nt) {
            const h16* bp = &Bs[buf][(wn * 32 + nt * 16 + lm) * BKT];
            v8h bl = *(const v8h*)(bp + lh * 8);
            v8h bh = *(const v8h*)(bp + 16 + lh * 8);
            bf[nt] = __builtin_shufflevector(bl, bh, 0,1,2,3,4,5,6,7,8,9,10,11,12,13,14,15);
        }
#pragma unroll
        for (int mt = 0; mt < 4; ++mt) {
            const h16* ap = &As[buf][(wm * 64 + mt * 16 + lm) * BKT];
            v8h lo = *(const v8h*)(ap + lh * 8);
            v8h hi = *(const v8h*)(ap + 16 + lh * 8);
            v16h af = __builtin_shufflevector(lo, hi, 0,1,2,3,4,5,6,7,8,9,10,11,12,13,14,15);
            acc[mt][0] = __builtin_amdgcn_wmma_f32_16x16x32_f16(false, af, false, bf[0], (short)0, acc[mt][0], false, false);
            acc[mt][1] = __builtin_amdgcn_wmma_f32_16x16x32_f16(false, af, false, bf[1], (short)0, acc[mt][1], false, false);
        }

        asm volatile("s_wait_asynccnt 0x0" ::: "memory");
        __syncthreads();
    }
#undef STAGE

    // ---- fused, branch-free epilogue ----
#pragma unroll
    for (int mt = 0; mt < 4; ++mt) {
#pragma unroll
        for (int nt = 0; nt < 2; ++nt) {
            int col = n0 + wn * 32 + nt * 16 + lm;
            float bv = (FLAGS & GF_BIAS) ? bias[col] : 0.f;
#pragma unroll
            for (int r = 0; r < 8; ++r) {
                int row = m0 + wm * 64 + mt * 16 + lh * 8 + r;
                float v = acc[mt][nt][r] + bv;
                if (FLAGS & GF_GELU) v = v / (1.f + __expf(-1.702f * v));  // fast GELU
                size_t o = (size_t)row * N + col;
                if (FLAGS & GF_RES) v += (float)resid[o];
                C[o] = (h16)v;
            }
        }
    }
}

// ---------------------------------------------------------------------------
// Weight convert + transpose: f32 W[K][N] -> f16 Wt[N][K]
// ---------------------------------------------------------------------------
__global__ void k_cvtw(const float* __restrict__ in, h16* __restrict__ out, int K, int N)
{
    int idx = blockIdx.x * blockDim.x + threadIdx.x;
    if (idx >= K * N) return;
    int n = idx / K, k = idx % K;
    out[idx] = (h16)in[(size_t)k * N + n];
}

// DFT matrices, packed: FwdT[128][512] = [cos ; -sin], InvT[512][128] = [sc*cos | -sc*sin]
__global__ void k_dftpack(h16* __restrict__ FwdT, h16* __restrict__ InvT)
{
    int i = blockIdx.x * blockDim.x + threadIdx.x;
    if (i >= LSEQ * NMOD) return;
    int l = i >> 6, m = i & 63;
    float ang = 6.28318530717958647f * (float)(l * m) / (float)LSEQ;
    float c = cosf(ang), s = sinf(ang);
    FwdT[(size_t)m * LSEQ + l] = (h16)c;
    FwdT[(size_t)(NMOD + m) * LSEQ + l] = (h16)(-s);
    float sc = ((m == 0) ? 1.f : 2.f) / (float)LSEQ;
    InvT[(size_t)l * 128 + m] = (h16)(sc * c);
    InvT[(size_t)l * 128 + 64 + m] = (h16)(-sc * s);
}

// ---------------------------------------------------------------------------
// Elementwise kernels (f16 activations, f32 math)
// ---------------------------------------------------------------------------
__global__ void k_embed(const float* __restrict__ x, const float* __restrict__ mark,
                        const float* __restrict__ vw, const float* __restrict__ tw,
                        h16* __restrict__ out, int Cin)
{
    int idx = blockIdx.x * blockDim.x + threadIdx.x;
    if (idx >= ROWS * DM) return;
    int b = idx / (LSEQ * DM);
    int r = idx - b * (LSEQ * DM);
    int l = r / DM, d = r % DM;
    int lmn = (l == 0) ? (LSEQ - 1) : (l - 1);
    int lpl = (l == LSEQ - 1) ? 0 : (l + 1);
    float s = 0.f;
    for (int c = 0; c < Cin; ++c) {
        const float* w = &vw[(d * Cin + c) * 3];
        s += w[0] * x[(b * LSEQ + lmn) * Cin + c]
           + w[1] * x[(b * LSEQ + l)   * Cin + c]
           + w[2] * x[(b * LSEQ + lpl) * Cin + c];
    }
    for (int t = 0; t < NTIME; ++t)
        s += mark[(b * LSEQ + l) * NTIME + t] * tw[t * DM + d];
    out[idx] = (h16)s;
}

__global__ void k_decomp_h(const h16* __restrict__ x, h16* __restrict__ season,
                           h16* __restrict__ trend, int acc)
{
    int idx = blockIdx.x * blockDim.x + threadIdx.x;
    if (idx >= ROWS * DM) return;
    int b = idx / (LSEQ * DM);
    int r = idx - b * (LSEQ * DM);
    int l = r / DM, d = r % DM;
    float s = 0.f;
    for (int j = -12; j <= 12; ++j) {
        int ll = l + j;
        ll = ll < 0 ? 0 : (ll > LSEQ - 1 ? LSEQ - 1 : ll);
        s += (float)x[((size_t)b * LSEQ + ll) * DM + d];
    }
    float mean = s * (1.f / 25.f);
    season[idx] = (h16)((float)x[idx] - mean);
    if (trend) trend[idx] = acc ? (h16)((float)trend[idx] + mean) : (h16)mean;
}

__global__ void k_decomp_f(const float* __restrict__ x, float* __restrict__ season,
                           float* __restrict__ trend)
{
    int idx = blockIdx.x * blockDim.x + threadIdx.x;
    if (idx >= NB * LSEQ * CINC) return;
    int b = idx / (LSEQ * CINC);
    int r = idx - b * (LSEQ * CINC);
    int l = r / CINC, c = r % CINC;
    float s = 0.f;
    for (int j = -12; j <= 12; ++j) {
        int ll = l + j;
        ll = ll < 0 ? 0 : (ll > LSEQ - 1 ? LSEQ - 1 : ll);
        s += x[(b * LSEQ + ll) * CINC + c];
    }
    float mean = s * (1.f / 25.f);
    season[idx] = x[idx] - mean;
    trend[idx] = mean;
}

__global__ void k_lnfeat(const h16* __restrict__ x, const float* __restrict__ w,
                         const float* __restrict__ bb, h16* __restrict__ y)
{
    int row = blockIdx.x * blockDim.x + threadIdx.x;
    if (row >= ROWS) return;
    const h16* xr = x + (size_t)row * DM;
    float mu = 0.f;
    for (int d = 0; d < DM; ++d) mu += (float)xr[d];
    mu *= (1.f / DM);
    float var = 0.f;
    for (int d = 0; d < DM; ++d) { float t = (float)xr[d] - mu; var += t * t; }
    var *= (1.f / DM);
    float inv = rsqrtf(var + LN_EPS);
    h16* yr = y + (size_t)row * DM;
    for (int d = 0; d < DM; ++d) yr[d] = (h16)(((float)xr[d] - mu) * inv * w[d] + bb[d]);
}

__global__ void k_subseqmean(h16* __restrict__ y)
{
    int idx = blockIdx.x * blockDim.x + threadIdx.x;  // (b,d)
    if (idx >= NB * DM) return;
    int b = idx / DM, d = idx % DM;
    float m = 0.f;
    for (int l = 0; l < LSEQ; ++l) m += (float)y[((size_t)b * LSEQ + l) * DM + d];
    m *= (1.f / LSEQ);
    for (int l = 0; l < LSEQ; ++l) {
        size_t o = ((size_t)b * LSEQ + l) * DM + d;
        y[o] = (h16)((float)y[o] - m);
    }
}

// (B,L,D) -> rows (b,h,e) x cols l
__global__ void k_perm(const h16* __restrict__ in, h16* __restrict__ out)
{
    int idx = blockIdx.x * blockDim.x + threadIdx.x;
    if (idx >= ROWS * DM) return;
    int b = idx / (LSEQ * DM);
    int r = idx - b * (LSEQ * DM);
    int l = r / DM, d = r % DM;
    int h = d >> 6, e = d & 63;
    out[((((size_t)b * NH + h) * DKH + e) << 9) + l] = in[idx];
}

// Y[b,h,o,(m|m+64)] = scale * sum_e X[b,h,e,(m|m+64)] (*) w[h,e,o,m]
__global__ void k_modemix(const h16* __restrict__ XF, const float* __restrict__ w,
                          h16* __restrict__ YF, float scale)
{
    int idx = blockIdx.x * blockDim.x + threadIdx.x;
    if (idx >= NB * NH * DKH * NMOD) return;
    int m = idx & 63;
    int o = (idx >> 6) & 63;
    int h = (idx >> 12) & 7;
    int b = idx >> 15;
    float ar = 0.f, ai = 0.f;
    for (int e = 0; e < DKH; ++e) {
        size_t xr0 = ((((size_t)b * NH + h) * DKH + e) << 7);
        float xr = (float)XF[xr0 + m], xi = (float)XF[xr0 + 64 + m];
        size_t wo = ((((((size_t)h << 6) + e) << 6) + o) << 6 | m) << 1;
        float wr = w[wo], wi = w[wo + 1];
        ar += xr * wr - xi * wi;
        ai += xr * wi + xi * wr;
    }
    size_t y0 = ((((size_t)b * NH + h) * DKH + o) << 7);
    YF[y0 + m]      = (h16)(ar * scale);
    YF[y0 + 64 + m] = (h16)(ai * scale);
}

// S[b,h,x,y] = tanh_c( sum_e Q[b,h,e,x] * K[b,h,e,y] )
__global__ void k_crossqk(const h16* __restrict__ QF, const h16* __restrict__ KF,
                          h16* __restrict__ Sr, h16* __restrict__ Si)
{
    int idx = blockIdx.x * blockDim.x + threadIdx.x;
    if (idx >= NB * NH * NMOD * NMOD) return;
    int y = idx & 63;
    int x = (idx >> 6) & 63;
    int h = (idx >> 12) & 7;
    int b = idx >> 15;
    size_t bh = (size_t)b * NH + h;
    float ar = 0.f, ai = 0.f;
    for (int e = 0; e < DKH; ++e) {
        size_t ro = ((bh * DKH + e) << 7);
        float qr = (float)QF[ro + x], qi = (float)QF[ro + 64 + x];
        float kr = (float)KF[ro + y], ki = (float)KF[ro + 64 + y];
        ar += qr * kr - qi * ki;
        ai += qr * ki + qi * kr;
    }
    float ta = tanhf(ar), tb = tanf(ai);
    float den = 1.f + ta * ta * tb * tb;
    size_t so = (bh << 12) + ((size_t)x << 6) + y;
    Sr[so] = (h16)(ta * (1.f + tb * tb) / den);
    Si[so] = (h16)(tb * (1.f - ta * ta) / den);
}

// V[b,h,e,(x|x+64)] = sum_y K[b,h,e,(y|y+64)] (*) S[b,h,x,y]
__global__ void k_crosskv(const h16* __restrict__ KF, const h16* __restrict__ Sr,
                          const h16* __restrict__ Si, h16* __restrict__ VF)
{
    int idx = blockIdx.x * blockDim.x + threadIdx.x;
    if (idx >= NB * NH * DKH * NMOD) return;
    int x = idx & 63;
    int e = (idx >> 6) & 63;
    int h = (idx >> 12) & 7;
    int b = idx >> 15;
    size_t bh = (size_t)b * NH + h;
    size_t ro = ((bh * DKH + e) << 7);
    float ar = 0.f, ai = 0.f;
    for (int y = 0; y < NMOD; ++y) {
        float kr = (float)KF[ro + y], ki = (float)KF[ro + 64 + y];
        size_t so = (bh << 12) + ((size_t)x << 6) + y;
        float sr = (float)Sr[so], si = (float)Si[so];
        ar += kr * sr - ki * si;
        ai += kr * si + ki * sr;
    }
    VF[ro + x]      = (h16)ar;
    VF[ro + 64 + x] = (h16)ai;
}

__global__ void k_trendinit(const float* __restrict__ xe, const float* __restrict__ sfull,
                            const float* __restrict__ tfull, float* __restrict__ seasinit,
                            float* __restrict__ trend7)
{
    int idx = blockIdx.x * blockDim.x + threadIdx.x;
    if (idx >= NB * LSEQ * CINC) return;
    int b = idx / (LSEQ * CINC);
    int r = idx - b * (LSEQ * CINC);
    int l = r / CINC, c = r % CINC;
    if (l < PRED) {
        seasinit[idx] = sfull[(b * LSEQ + 256 + l) * CINC + c];
        trend7[idx]   = tfull[(b * LSEQ + 256 + l) * CINC + c];
    } else {
        seasinit[idx] = 0.f;
        float m = 0.f;
        for (int ll = 0; ll < LSEQ; ++ll) m += xe[(b * LSEQ + ll) * CINC + c];
        trend7[idx] = m * (1.f / LSEQ);
    }
}

__global__ void k_trendconv(const h16* __restrict__ t, const float* __restrict__ w,
                            float* __restrict__ trend7)
{
    int idx = blockIdx.x * blockDim.x + threadIdx.x;  // (b,l,o)
    if (idx >= NB * LSEQ * CINC) return;
    int b = idx / (LSEQ * CINC);
    int r = idx - b * (LSEQ * CINC);
    int l = r / CINC, o = r % CINC;
    int lmn = (l == 0) ? (LSEQ - 1) : (l - 1);
    int lpl = (l == LSEQ - 1) ? 0 : (l + 1);
    float s = 0.f;
    for (int c = 0; c < DM; ++c) {
        const float* wp = &w[(o * DM + c) * 3];
        s += wp[0] * (float)t[((size_t)b * LSEQ + lmn) * DM + c]
           + wp[1] * (float)t[((size_t)b * LSEQ + l)   * DM + c]
           + wp[2] * (float)t[((size_t)b * LSEQ + lpl) * DM + c];
    }
    trend7[idx] += s;
}

__global__ void k_final(const h16* __restrict__ xln, const float* __restrict__ pw,
                        const float* __restrict__ pb, const float* __restrict__ trend7,
                        float* __restrict__ out)
{
    int idx = blockIdx.x * blockDim.x + threadIdx.x;  // (b,t,c)
    if (idx >= NB * PRED * CINC) return;
    int b = idx / (PRED * CINC);
    int r = idx - b * (PRED * CINC);
    int t = r / CINC, c = r % CINC;
    int l = 256 + t;
    float s = pb[c];
    const h16* xr = xln + ((size_t)b * LSEQ + l) * DM;
    for (int d = 0; d < DM; ++d) s += (float)xr[d] * pw[d * CINC + c];
    out[idx] = s + trend7[(b * LSEQ + l) * CINC + c];
}

// ---------------------------------------------------------------------------
// Host orchestration
// ---------------------------------------------------------------------------
static inline void gemm(const h16* A, const h16* Bt, h16* C, const float* bias,
                        const h16* resid, int M, int N, int K, int flags, hipStream_t s)
{
    dim3 g(N / BNT, M / BMT);
    dim3 b(256);
    switch (flags) {
    case 0:                 k_wmma_gemm<0>                <<<g, b, 0, s>>>(A, Bt, C, bias, resid, M, N, K); break;
    case GF_BIAS:           k_wmma_gemm<GF_BIAS>          <<<g, b, 0, s>>>(A, Bt, C, bias, resid, M, N, K); break;
    case GF_GELU:           k_wmma_gemm<GF_GELU>          <<<g, b, 0, s>>>(A, Bt, C, bias, resid, M, N, K); break;
    case GF_RES:            k_wmma_gemm<GF_RES>           <<<g, b, 0, s>>>(A, Bt, C, bias, resid, M, N, K); break;
    default:                k_wmma_gemm<GF_BIAS | GF_RES> <<<g, b, 0, s>>>(A, Bt, C, bias, resid, M, N, K); break;
    }
}

#define EL(kern, total, ...) kern<<<((total) + 255) / 256, 256, 0, stream>>>(__VA_ARGS__)

extern "C" void kernel_launch(void* const* d_in, const int* in_sizes, int n_in,
                              void* d_out, int out_size, void* d_ws, size_t ws_size,
                              hipStream_t stream)
{
    (void)in_sizes; (void)n_in; (void)out_size; (void)ws_size;
    const float* x_enc  = (const float*)d_in[0];
    const float* mark_e = (const float*)d_in[1];
    const float* mark_d = (const float*)d_in[3];   // x_dec values unused by the model
    const float* enc_vw = (const float*)d_in[4];
    const float* enc_tw = (const float*)d_in[5];
    const float* dec_vw = (const float*)d_in[6];
    const float* dec_tw = (const float*)d_in[7];
    const float* enc_fb = (const float*)d_in[8];
    const float* dec_fb = (const float*)d_in[9];
    const float* crs_fb = (const float*)d_in[10];
    const float* enc_ln_w = (const float*)d_in[31];
    const float* enc_ln_b = (const float*)d_in[32];
    const float* dec_ln_w = (const float*)d_in[52];
    const float* dec_ln_b = (const float*)d_in[53];
    const float* proj_w   = (const float*)d_in[54];
    const float* proj_b   = (const float*)d_in[55];
    float* out = (float*)d_out;

    char* base = (char*)d_ws; size_t off = 0;
    auto ALLOCH = [&](size_t n) -> h16* {
        h16* p = reinterpret_cast<h16*>(base + off);
        off += ((n * sizeof(h16)) + 255) & ~(size_t)255;
        return p;
    };
    auto ALLOCF = [&](size_t n) -> float* {
        float* p = reinterpret_cast<float*>(base + off);
        off += ((n * sizeof(float)) + 255) & ~(size_t)255;
        return p;
    };

    h16* FwdT = ALLOCH((size_t)128 * LSEQ);
    h16* InvT = ALLOCH((size_t)LSEQ * 128);
    h16* wt_eq[2]; h16* wt_eo[2]; h16* wt_ec1[2]; h16* wt_ec2[2];
    for (int li = 0; li < 2; ++li) {
        wt_eq[li]  = ALLOCH((size_t)DM * DM);
        wt_eo[li]  = ALLOCH((size_t)DM * DM);
        wt_ec1[li] = ALLOCH((size_t)DM * DFF);
        wt_ec2[li] = ALLOCH((size_t)DFF * DM);
    }
    h16* wt_sq  = ALLOCH((size_t)DM * DM);
    h16* wt_so  = ALLOCH((size_t)DM * DM);
    h16* wt_cq  = ALLOCH((size_t)DM * DM);
    h16* wt_ck  = ALLOCH((size_t)DM * DM);
    h16* wt_co  = ALLOCH((size_t)DM * DM);
    h16* wt_dc1 = ALLOCH((size_t)DM * DFF);
    h16* wt_dc2 = ALLOCH((size_t)DFF * DM);
    h16* encx   = ALLOCH((size_t)ROWS * DM);
    h16* newx   = ALLOCH((size_t)ROWS * DM);
    h16* q      = ALLOCH((size_t)ROWS * DM);
    h16* qperm  = ALLOCH((size_t)ROWS * DM);
    h16* tdom   = ALLOCH((size_t)ROWS * DM);
    h16* encout = ALLOCH((size_t)ROWS * DM);
    h16* decx   = ALLOCH((size_t)ROWS * DM);
    h16* xln    = ALLOCH((size_t)ROWS * DM);
    h16* tsum   = ALLOCH((size_t)ROWS * DM);
    h16* ffnmid = ALLOCH((size_t)ROWS * DFF);
    h16* XF = ALLOCH((size_t)ROWS * 128);
    h16* YF = ALLOCH((size_t)ROWS * 128);
    h16* KF = ALLOCH((size_t)ROWS * 128);
    h16* VF = ALLOCH((size_t)ROWS * 128);
    h16* Sr = ALLOCH((size_t)NB * NH * NMOD * NMOD);
    h16* Si = ALLOCH((size_t)NB * NH * NMOD * NMOD);
    float* sfull    = ALLOCF((size_t)NB * LSEQ * CINC);
    float* tfull    = ALLOCF((size_t)NB * LSEQ * CINC);
    float* seasinit = ALLOCF((size_t)NB * LSEQ * CINC);
    float* trend7   = ALLOCF((size_t)NB * LSEQ * CINC);

    // ---- constants / weight prep ----
    EL(k_dftpack, LSEQ * NMOD, FwdT, InvT);
    for (int li = 0; li < 2; ++li) {
        EL(k_cvtw, DM * DM,  (const float*)d_in[11 + li * 10 + 0], wt_eq[li],  DM, DM);
        EL(k_cvtw, DM * DM,  (const float*)d_in[11 + li * 10 + 6], wt_eo[li],  DM, DM);
        EL(k_cvtw, DM * DFF, (const float*)d_in[11 + li * 10 + 8], wt_ec1[li], DM, DFF);
        EL(k_cvtw, DFF * DM, (const float*)d_in[11 + li * 10 + 9], wt_ec2[li], DFF, DM);
    }
    EL(k_cvtw, DM * DM,  (const float*)d_in[33], wt_sq, DM, DM);
    EL(k_cvtw, DM * DM,  (const float*)d_in[39], wt_so, DM, DM);
    EL(k_cvtw, DM * DM,  (const float*)d_in[41], wt_cq, DM, DM);
    EL(k_cvtw, DM * DM,  (const float*)d_in[43], wt_ck, DM, DM);
    EL(k_cvtw, DM * DM,  (const float*)d_in[47], wt_co, DM, DM);
    EL(k_cvtw, DM * DFF, (const float*)d_in[49], wt_dc1, DM, DFF);
    EL(k_cvtw, DFF * DM, (const float*)d_in[50], wt_dc2, DFF, DM);

    // ===== encoder =====
    EL(k_embed, ROWS * DM, x_enc, mark_e, enc_vw, enc_tw, encx, CINC);
    for (int li = 0; li < 2; ++li) {
        const float* qb = (const float*)d_in[11 + li * 10 + 1];
        const float* ob = (const float*)d_in[11 + li * 10 + 7];
        gemm(encx, wt_eq[li], q, qb, nullptr, ROWS, DM, DM, GF_BIAS, stream);
        EL(k_perm, ROWS * DM, q, qperm);
        gemm(qperm, FwdT, XF, nullptr, nullptr, ROWS, 128, DM, 0, stream);
        EL(k_modemix, NB * NH * DKH * NMOD, XF, enc_fb, YF, 1.f);
        gemm(YF, InvT, tdom, nullptr, nullptr, ROWS, DM, 128, 0, stream);
        gemm(tdom, wt_eo[li], newx, ob, encx, ROWS, DM, DM, GF_BIAS | GF_RES, stream);
        EL(k_decomp_h, ROWS * DM, newx, encx, (h16*)nullptr, 0);
        gemm(encx, wt_ec1[li], ffnmid, nullptr, nullptr, ROWS, DFF, DM, GF_GELU, stream);
        gemm(ffnmid, wt_ec2[li], newx, nullptr, encx, ROWS, DM, DFF, GF_RES, stream);
        EL(k_decomp_h, ROWS * DM, newx, encx, (h16*)nullptr, 0);
    }
    EL(k_lnfeat, ROWS, encx, enc_ln_w, enc_ln_b, encout);
    EL(k_subseqmean, NB * DM, encout);

    // ===== decoder init =====
    EL(k_decomp_f, NB * LSEQ * CINC, x_enc, sfull, tfull);
    EL(k_trendinit, NB * LSEQ * CINC, x_enc, sfull, tfull, seasinit, trend7);
    EL(k_embed, ROWS * DM, seasinit, mark_d, dec_vw, dec_tw, decx, CINC);

    // ===== decoder layer 0 =====
    const float* sqb = (const float*)d_in[34];
    const float* sob = (const float*)d_in[40];
    const float* cqb = (const float*)d_in[42];
    const float* ckb = (const float*)d_in[44];
    const float* cob = (const float*)d_in[48];
    const float* trw = (const float*)d_in[51];

    // self (Fourier block)
    gemm(decx, wt_sq, q, sqb, nullptr, ROWS, DM, DM, GF_BIAS, stream);
    EL(k_perm, ROWS * DM, q, qperm);
    gemm(qperm, FwdT, XF, nullptr, nullptr, ROWS, 128, DM, 0, stream);
    EL(k_modemix, NB * NH * DKH * NMOD, XF, dec_fb, YF, 1.f);
    gemm(YF, InvT, tdom, nullptr, nullptr, ROWS, DM, 128, 0, stream);
    gemm(tdom, wt_so, newx, sob, decx, ROWS, DM, DM, GF_BIAS | GF_RES, stream);
    EL(k_decomp_h, ROWS * DM, newx, decx, tsum, 0);      // t1

    // cross (Fourier cross attention; k doubles as v)
    gemm(decx, wt_cq, q, cqb, nullptr, ROWS, DM, DM, GF_BIAS, stream);
    EL(k_perm, ROWS * DM, q, qperm);
    gemm(qperm, FwdT, XF, nullptr, nullptr, ROWS, 128, DM, 0, stream);   // Q_ft
    gemm(encout, wt_ck, q, ckb, nullptr, ROWS, DM, DM, GF_BIAS, stream);
    EL(k_perm, ROWS * DM, q, qperm);
    gemm(qperm, FwdT, KF, nullptr, nullptr, ROWS, 128, DM, 0, stream);   // K_ft
    EL(k_crossqk, NB * NH * NMOD * NMOD, XF, KF, Sr, Si);
    EL(k_crosskv, NB * NH * DKH * NMOD, KF, Sr, Si, VF);
    EL(k_modemix, NB * NH * DKH * NMOD, VF, crs_fb, YF, 1.f / ((float)DM * (float)DM));
    gemm(YF, InvT, tdom, nullptr, nullptr, ROWS, DM, 128, 0, stream);
    gemm(tdom, wt_co, newx, cob, decx, ROWS, DM, DM, GF_BIAS | GF_RES, stream);
    EL(k_decomp_h, ROWS * DM, newx, decx, tsum, 1);      // t2 (accumulate)

    // FFN
    gemm(decx, wt_dc1, ffnmid, nullptr, nullptr, ROWS, DFF, DM, GF_GELU, stream);
    gemm(ffnmid, wt_dc2, newx, nullptr, decx, ROWS, DM, DFF, GF_RES, stream);
    EL(k_decomp_h, ROWS * DM, newx, decx, tsum, 1);      // t3 (accumulate)

    EL(k_trendconv, NB * LSEQ * CINC, tsum, trw, trend7);

    // ===== output =====
    EL(k_lnfeat, ROWS, decx, dec_ln_w, dec_ln_b, xln);
    EL(k_subseqmean, NB * DM, xln);
    EL(k_final, NB * PRED * CINC, xln, proj_w, proj_b, trend7, out);
}
